// DebateGraph_42958262894976
// MI455X (gfx1250) — matrix operand order repaired and checked by reading, not compiled, not run on hardware
//
#include <hip/hip_runtime.h>

#define NFEAT 256
#define NHID  256
#define NHEADS 4
#define DH    64
#define LRALPHA 0.2f
#define BB 16
#define TT 6
#define NN 128
#define NTOT (BB*TT*NN)      /* 12288 */
#define EDG  (NTOT*16)       /* 196608 */

typedef __attribute__((ext_vector_type(16))) _Float16 v16h;
typedef __attribute__((ext_vector_type(8)))  _Float16 v8h;
typedef __attribute__((ext_vector_type(8)))  float    v8f;

// pack f32 weight [K x N] row-major into fragment-native f16 layout:
// Bp[(((nt*KT)+kt)*32 + lane)*16 + e]  with  n = nt*16+(lane&15),
// k = kt*32 + 8*(lane>=16) + (e&7) + 16*(e>>3)   (CDNA5 B-fragment layout, 16x16x32)
__global__ void k_pack_b(const float* __restrict__ B, _Float16* __restrict__ Bp,
                         int K, int N) {
    int idx = blockIdx.x * blockDim.x + threadIdx.x;     // K*N threads
    if (idx >= K * N) return;
    int e    = idx & 15;
    int lane = (idx >> 4) & 31;
    int tile = idx >> 9;                                 // nt*KT + kt
    int KT = K >> 5;
    int nt = tile / KT, kt = tile % KT;
    int n = nt * 16 + (lane & 15);
    int k = kt * 32 + ((lane >> 4) << 3) + (e & 7) + ((e >> 3) << 4);
    Bp[idx] = (_Float16)B[(size_t)k * N + n];
}

// one K-pass of the 16x64 strip, software-pipelined (double-buffered frags)
template <int KT>
__device__ __forceinline__ void gemm_pass(const _Float16* __restrict__ aB,
                                          const _Float16* __restrict__ bB,
                                          v8f acc[4]) {
    union AV { v16h v; v8h h[2]; };
    AV a_cur, a_nxt;
    v16h b_cur[4], b_nxt[4];
    a_cur.h[0] = *(const v8h*)(aB);
    a_cur.h[1] = *(const v8h*)(aB + 16);
#pragma unroll
    for (int j = 0; j < 4; ++j) b_cur[j] = *(const v16h*)(bB + j * KT * 512);
#pragma unroll
    for (int kt = 0; kt < KT; ++kt) {
        if (kt + 1 < KT) {          // issue next iteration's loads before the WMMAs
            a_nxt.h[0] = *(const v8h*)(aB + (kt + 1) * 32);
            a_nxt.h[1] = *(const v8h*)(aB + (kt + 1) * 32 + 16);
#pragma unroll
            for (int j = 0; j < 4; ++j)
                b_nxt[j] = *(const v16h*)(bB + (j * KT + kt + 1) * 512);
        }
#pragma unroll
        for (int j = 0; j < 4; ++j)
            acc[j] = __builtin_amdgcn_wmma_f32_16x16x32_f16(false, a_cur.v, false, b_cur[j],
                                                            (short)0, acc[j], false, false);
        if (kt + 1 < KT) {
            a_cur = a_nxt;
#pragma unroll
            for (int j = 0; j < 4; ++j) b_cur[j] = b_nxt[j];
        }
    }
}

// ---- fused dual-GEMM: C = act(A1@B1 [+ A2@B2] [+ bias]) -------------------
// K compile-time; ACT: 0=identity, 1=sigmoid, 2=tanh. DUAL: second A@B pair.
// One wave per 16x64 output strip (4 n-tiles); all fragment addresses are a
// per-lane base pointer + compile-time byte offset.
template <int K, int ACT, bool DUAL>
__global__ __launch_bounds__(256)
void k_gemm2(const _Float16* __restrict__ A1, const _Float16* __restrict__ Bp1,
             const _Float16* __restrict__ A2, const _Float16* __restrict__ Bp2,
             const float* __restrict__ bias, float* __restrict__ C,
             int M, int N) {
    constexpr int KT = K / 32;
    int wave = (blockIdx.x * blockDim.x + threadIdx.x) >> 5;
    int lane = threadIdx.x & 31;
    int ngrp = N >> 6;                                   // groups of 4 n-tiles
    int mt = wave / ngrp, ng = wave % ngrp;
    int m0 = mt * 16;
    if (m0 >= M) return;                                 // wave-uniform
    int nt0 = ng * 4;

    // per-lane bases (all further offsets are compile-time constants)
    const int arow = (m0 + (lane & 15)) * K + ((lane >> 4) << 3);
    const int brow = (nt0 * KT * 32 + lane) * 16;

    v8f acc[4];
#pragma unroll
    for (int j = 0; j < 4; ++j) acc[j] = (v8f){0.f,0.f,0.f,0.f,0.f,0.f,0.f,0.f};

    gemm_pass<KT>(A1 + arow, Bp1 + brow, acc);
    if (DUAL) gemm_pass<KT>(A2 + arow, Bp2 + brow, acc);

    // epilogue: per-lane base + constant offsets
    int half = lane >> 4;
    int ncol = nt0 * 16 + (lane & 15);
    float* cB = C + (size_t)(m0 + 8 * half) * N + ncol;
#pragma unroll
    for (int j = 0; j < 4; ++j) {
        float bv = bias ? bias[ncol + j * 16] : 0.f;
#pragma unroll
        for (int v = 0; v < 8; ++v) {
            float x = acc[j][v] + bv;
            if (ACT == 1)      x = 1.f / (1.f + __expf(-x));
            else if (ACT == 2) x = tanhf(x);
            cB[v * N + j * 16] = x;
        }
    }
}

// ---- small elementwise / graph kernels -----------------------------------

__global__ void k_f2h(const float* __restrict__ x, _Float16* __restrict__ y, int n) {
    int i = blockIdx.x * blockDim.x + threadIdx.x;
    if (i < n) y[i] = (_Float16)x[i];
}

__global__ void k_init_h(const float* __restrict__ emb, const float* __restrict__ nmask,
                         const int* __restrict__ nodeidx, float* __restrict__ h) {
    int idx = blockIdx.x * blockDim.x + threadIdx.x;       // NTOT*NFEAT
    if (idx >= NTOT * NFEAT) return;
    int i = idx >> 8, c = idx & 255;
    int s = nodeidx[i];
    h[idx] = emb[(size_t)s * NFEAT + c] * nmask[s];
}

__global__ void k_attn_coef(const float* __restrict__ Wh,
                            const float* __restrict__ a_src, const float* __restrict__ a_dst,
                            float* __restrict__ es, float* __restrict__ ed) {
    int idx = blockIdx.x * blockDim.x + threadIdx.x;       // NTOT*NHEADS
    if (idx >= NTOT * NHEADS) return;
    int i = idx >> 2, hd = idx & 3;
    const float* w  = Wh + (size_t)i * NHID + hd * DH;
    const float* as = a_src + hd * DH;
    const float* ad = a_dst + hd * DH;
    float s = 0.f, d = 0.f;
#pragma unroll 8
    for (int j = 0; j < DH; ++j) { float x = w[j]; s += x * as[j]; d += x * ad[j]; }
    es[idx] = s; ed[idx] = d;
}

__global__ void k_edge_w(const float* __restrict__ es, const float* __restrict__ ed,
                         const int* __restrict__ src, const int* __restrict__ dst,
                         float* __restrict__ wbuf, float* __restrict__ den) {
    int idx = blockIdx.x * blockDim.x + threadIdx.x;       // EDG*NHEADS
    if (idx >= EDG * NHEADS) return;
    int e = idx >> 2, hd = idx & 3;
    float l = es[src[e]*NHEADS + hd] + ed[dst[e]*NHEADS + hd];
    l = (l > 0.f) ? l : LRALPHA * l;                       // leaky_relu; softmax is
    float w = __expf(l);                                   // shift-invariant -> skip max
    wbuf[idx] = w;
    atomicAdd(&den[dst[e]*NHEADS + hd], w);
}

// one wave per edge: attn = w/(den+eps); msg[dst] += attn * Wh[src]
__global__ __launch_bounds__(256)
void k_edge_msg(const float* __restrict__ Wh, const float* __restrict__ wbuf,
                const float* __restrict__ den, const int* __restrict__ src,
                const int* __restrict__ dst, float* __restrict__ attn,
                float* __restrict__ msg) {
    int e    = (blockIdx.x * blockDim.x + threadIdx.x) >> 5;
    int lane = threadIdx.x & 31;
    if (e >= EDG) return;
    int s = src[e], d = dst[e];
    int hd = lane >> 3;                                    // 8 lanes per head
    float a = wbuf[e*NHEADS + hd] / (den[d*NHEADS + hd] + 1e-9f);
    if (lane < NHEADS)
        attn[e*NHEADS + lane] = wbuf[e*NHEADS + lane] / (den[d*NHEADS + lane] + 1e-9f);
    const float* wrow = Wh  + (size_t)s * NHID + lane * 8;
    float*       mrow = msg + (size_t)d * NHID + lane * 8;
#pragma unroll
    for (int j = 0; j < 8; ++j) atomicAdd(&mrow[j], a * wrow[j]);
}

__global__ void k_rh(const float* __restrict__ r, const float* __restrict__ h,
                     _Float16* __restrict__ rh) {
    int i = blockIdx.x * blockDim.x + threadIdx.x;
    if (i < NTOT * NHID) rh[i] = (_Float16)(r[i] * h[i]);
}

__global__ void k_update(float* __restrict__ h, const float* __restrict__ z,
                         const float* __restrict__ n, const int* __restrict__ tid, int t) {
    int idx = blockIdx.x * blockDim.x + threadIdx.x;
    if (idx >= NTOT * NHID) return;
    if (tid[idx >> 8] == t) {
        float zz = z[idx];
        h[idx] = (1.f - zz) * h[idx] + zz * n[idx];
    }
}

__global__ void k_score(const float* __restrict__ attn, const int* __restrict__ dst,
                        float* __restrict__ score) {
    int e = blockIdx.x * blockDim.x + threadIdx.x;
    if (e >= EDG) return;
    float m = 0.25f * (attn[e*4] + attn[e*4+1] + attn[e*4+2] + attn[e*4+3]);
    atomicAdd(&score[dst[e]], m);
}

// block per (b,t): softmax over N=128 nodes, pooled mean, accumulate speaker sums
__global__ __launch_bounds__(256)
void k_readout(const float* __restrict__ score, const float* __restrict__ h,
               float* __restrict__ out) {
    __shared__ float s[NN];
    int b = blockIdx.x / TT, t = blockIdx.x % TT;
    int nbase = (b * TT + t) * NN;
    if (threadIdx.x < NN) s[threadIdx.x] = score[nbase + threadIdx.x];
    __syncthreads();
    float mx = -1e30f;
    for (int n = 0; n < NN; ++n) mx = fmaxf(mx, s[n]);
    float den = 0.f;
    for (int n = 0; n < NN; ++n) den += __expf(s[n] - mx);
    float inv = 1.f / den;
    int c = threadIdx.x;                                   // 256 == NHID
    float acc = 0.f;
    for (int n = 0; n < NN; ++n)
        acc += __expf(s[n] - mx) * inv * h[(size_t)(nbase + n) * NHID + c];
    acc *= (1.f / NN);
    atomicAdd(&out[(t & 1) * (BB * NHID) + b * NHID + c], acc);
}

// ---- orchestration --------------------------------------------------------

extern "C" void kernel_launch(void* const* d_in, const int* in_sizes, int n_in,
                              void* d_out, int out_size, void* d_ws, size_t ws_size,
                              hipStream_t stream) {
    const float* emb     = (const float*)d_in[0];
    const float* nmask   = (const float*)d_in[1];
    const float* W       = (const float*)d_in[2];
    const float* a_src   = (const float*)d_in[3];
    const float* a_dst   = (const float*)d_in[4];
    const float* Wz      = (const float*)d_in[5];
    const float* Uz      = (const float*)d_in[6];
    const float* Wr      = (const float*)d_in[7];
    const float* Ur      = (const float*)d_in[8];
    const float* Wn      = (const float*)d_in[9];
    const float* Un      = (const float*)d_in[10];
    const float* bz      = (const float*)d_in[11];
    const float* br      = (const float*)d_in[12];
    const float* bn      = (const float*)d_in[13];
    const int*   nodeidx = (const int*)d_in[14];
    const int*   esrc    = (const int*)d_in[15];
    const int*   edst    = (const int*)d_in[16];
    const int*   tid     = (const int*)d_in[17];
    float* out = (float*)d_out;

    char* base = (char*)d_ws; size_t off = 0;
    auto alloc = [&](size_t bytes) -> void* {
        void* p = base + off; off += (bytes + 255) & ~(size_t)255; return p;
    };
    const size_t MATF = (size_t)NTOT * NHID * sizeof(float);
    const size_t MATH = (size_t)NTOT * NHID * sizeof(_Float16);
    float*    h     = (float*)alloc(MATF);
    float*    Wh    = (float*)alloc(MATF);
    float*    msg   = (float*)alloc(MATF);
    float*    zbuf  = (float*)alloc(MATF);
    float*    rbuf  = (float*)alloc(MATF);
    float*    nbuf  = (float*)alloc(MATF);
    _Float16* h16   = (_Float16*)alloc(MATH);
    _Float16* msg16 = (_Float16*)alloc(MATH);
    _Float16* rh16  = (_Float16*)alloc(MATH);
    const size_t WB = (size_t)NHID * NHID * sizeof(_Float16);
    _Float16* Wp  = (_Float16*)alloc(WB);
    _Float16* Wzp = (_Float16*)alloc(WB);
    _Float16* Uzp = (_Float16*)alloc(WB);
    _Float16* Wrp = (_Float16*)alloc(WB);
    _Float16* Urp = (_Float16*)alloc(WB);
    _Float16* Wnp = (_Float16*)alloc(WB);
    _Float16* Unp = (_Float16*)alloc(WB);
    float* es    = (float*)alloc((size_t)NTOT * NHEADS * sizeof(float));
    float* ed    = (float*)alloc((size_t)NTOT * NHEADS * sizeof(float));
    float* den   = (float*)alloc((size_t)NTOT * NHEADS * sizeof(float));
    float* wb    = (float*)alloc((size_t)EDG * NHEADS * sizeof(float));
    float* attn  = (float*)alloc((size_t)EDG * NHEADS * sizeof(float));
    float* score = (float*)alloc((size_t)NTOT * sizeof(float));
    (void)ws_size; (void)in_sizes; (void)n_in;

    const int EW  = NHID * NHID;                 // 65536 weight elems
    const int ELT = NTOT * NHID;                 // 3.1M elems
    // one wave per 16x64 strip: waves = (M/16)*(N/64), 8 waves/block
    const int GEMM_BLKS = ((NTOT / 16) * (NHID / 64)) / 8;

    // one-time (per call) prep: pack weights into fragment-native f16 layout
    k_pack_b<<<EW/256, 256, 0, stream>>>(W,  Wp,  NFEAT, NHID);
    k_pack_b<<<EW/256, 256, 0, stream>>>(Wz, Wzp, NHID, NHID);
    k_pack_b<<<EW/256, 256, 0, stream>>>(Uz, Uzp, NHID, NHID);
    k_pack_b<<<EW/256, 256, 0, stream>>>(Wr, Wrp, NHID, NHID);
    k_pack_b<<<EW/256, 256, 0, stream>>>(Ur, Urp, NHID, NHID);
    k_pack_b<<<EW/256, 256, 0, stream>>>(Wn, Wnp, NHID, NHID);
    k_pack_b<<<EW/256, 256, 0, stream>>>(Un, Unp, NHID, NHID);
    k_init_h<<<ELT/256, 256, 0, stream>>>(emb, nmask, nodeidx, h);
    hipMemsetAsync(out,   0, (size_t)out_size * sizeof(float), stream);
    hipMemsetAsync(score, 0, (size_t)NTOT * sizeof(float), stream);

    for (int t = 0; t < TT; ++t) {
        k_f2h<<<ELT/256, 256, 0, stream>>>(h, h16, ELT);
        // Wh = h @ W
        k_gemm2<NFEAT, 0, false><<<GEMM_BLKS, 256, 0, stream>>>(
            h16, Wp, (const _Float16*)nullptr, (const _Float16*)nullptr,
            (const float*)nullptr, Wh, NTOT, NHID);
        k_attn_coef<<<(NTOT*NHEADS)/256, 256, 0, stream>>>(Wh, a_src, a_dst, es, ed);
        hipMemsetAsync(den, 0, (size_t)NTOT * NHEADS * sizeof(float), stream);
        hipMemsetAsync(msg, 0, MATF, stream);
        k_edge_w<<<(EDG*NHEADS)/256, 256, 0, stream>>>(es, ed, esrc, edst, wb, den);
        k_edge_msg<<<EDG/8, 256, 0, stream>>>(Wh, wb, den, esrc, edst, attn, msg);
        k_f2h<<<ELT/256, 256, 0, stream>>>(msg, msg16, ELT);
        // z = sigmoid(msg@Wz + h@Uz + bz) ; r = sigmoid(msg@Wr + h@Ur + br)
        k_gemm2<NHID, 1, true><<<GEMM_BLKS, 256, 0, stream>>>(msg16, Wzp, h16, Uzp,
                                                              bz, zbuf, NTOT, NHID);
        k_gemm2<NHID, 1, true><<<GEMM_BLKS, 256, 0, stream>>>(msg16, Wrp, h16, Urp,
                                                              br, rbuf, NTOT, NHID);
        k_rh<<<ELT/256, 256, 0, stream>>>(rbuf, h, rh16);
        // n = tanh(msg@Wn + (r*h)@Un + bn)
        k_gemm2<NHID, 2, true><<<GEMM_BLKS, 256, 0, stream>>>(msg16, Wnp, rh16, Unp,
                                                              bn, nbuf, NTOT, NHID);
        k_update<<<ELT/256, 256, 0, stream>>>(h, zbuf, nbuf, tid, t);
    }

    k_score<<<EDG/256, 256, 0, stream>>>(attn, edst, score);
    k_readout<<<BB*TT, 256, 0, stream>>>(score, h, out);
}
